// Attention_55327768708220
// MI455X (gfx1250) — matrix operand order repaired and checked by reading, not compiled, not run on hardware
//
#include <hip/hip_runtime.h>
#include <hip/hip_bf16.h>

// ---------------------------------------------------------------------------
// Fused attention (BERT-style), MI455X / gfx1250, wave32.
//   B=2, F=T=2048, D=1024, N=16 heads, H=64.
// All GEMMs on V_WMMA_F32_16X16X32_BF16 (f32 accumulate).
// Pre-passes convert activations to bf16 and transpose weights so the GEMM
// kernels issue only contiguous b128 fragment loads (192MB L2 carries reuse).
// Attention K/Vt tiles are DMA'd to LDS by the Tensor Data Mover with
// double-buffering (DMA of tile i+1 overlaps WMMA+softmax of tile i),
// falling back to GLOBAL_LOAD_ASYNC_TO_LDS_B128, then plain copies.
// ---------------------------------------------------------------------------

#define B_   2
#define S_   2048        // F == T
#define D_   1024
#define NH_  16          // heads
#define H_   64          // head dim
#define NC_  (NH_ * H_)  // 1024 output columns per projection

typedef __attribute__((ext_vector_type(16))) __bf16 v16bf;
typedef __attribute__((ext_vector_type(8)))  __bf16 v8bf;
typedef __attribute__((ext_vector_type(8)))  float  v8f;
typedef __attribute__((ext_vector_type(4)))  int    v4i;
typedef __attribute__((ext_vector_type(8)))  int    v8i;
typedef __attribute__((ext_vector_type(4)))  unsigned int v4u;

union ABfrag { v16bf v; v8bf h[2]; };

#ifndef __has_builtin
#define __has_builtin(x) 0
#endif

#if __has_builtin(__builtin_amdgcn_global_load_async_to_lds_b128)
#define HAVE_ASYNC_LDS 1
#else
#warning "CDNA5-probe: __builtin_amdgcn_global_load_async_to_lds_b128 ABSENT"
#endif
#if __has_builtin(__builtin_amdgcn_s_wait_asynccnt)
#define HAVE_WAIT_ASYNC 1
#endif
#if __has_builtin(__builtin_amdgcn_tensor_load_to_lds) && \
    __has_builtin(__builtin_amdgcn_s_wait_tensorcnt) &&   \
    __has_include(<hip/amd_detail/amd_gfx1250_TDM.h>)
#define HAVE_TDM 1        // amdgpu-toolchain: 6-arg tensor_load_to_lds
#else
#warning "CDNA5-probe: TDM path ABSENT"
#endif

// --- 16-byte global -> LDS copy; async (ASYNCcnt) when available ----------
__device__ __forceinline__ void copy16_g2l(const void* g, void* l) {
#if defined(HAVE_ASYNC_LDS)
    __builtin_amdgcn_global_load_async_to_lds_b128(
        (__attribute__((address_space(1))) v4i*)(unsigned long long)g,
        (__attribute__((address_space(3))) v4i*)(unsigned int)(unsigned long long)l,
        0, 0);
#else
    *(int4*)l = *(const int4*)g;
#endif
}

#if defined(HAVE_TDM)
// --- TDM 2D tile load: one instruction moves a [d1 x d0] bf16 tile --------
// D# group0/group1 packed per CDNA5 ISA 8.3/8.4 (dims/strides in elements,
// data_size=1 => 2-byte elements, descriptor type=2 "image").
__device__ __forceinline__ void tdm_load_2d(const void* gaddr, void* lds,
                                            unsigned d0, unsigned d1,
                                            unsigned stride0) {
    const unsigned long long ga = (unsigned long long)gaddr;
    const unsigned la = (unsigned)(unsigned long long)lds;
    v4u g0;
    g0[0] = 1u;                                   // count=1 (valid user D#)
    g0[1] = la;                                   // lds_addr (bytes)
    g0[2] = (unsigned)ga;                         // global_addr[31:0]
    g0[3] = (unsigned)(ga >> 32) | 0x80000000u;   // global_addr[56:32] | type=2
    v8i g1;
    g1[0] = (int)(1u << 16);                      // data_size = 2 bytes
    g1[1] = (int)((d0 & 0xffffu) << 16);          // tensor_dim0 lo16 @ bits 63:48
    g1[2] = (int)(((d0 >> 16) & 0xffffu) | ((d1 & 0xffffu) << 16)); // td0 hi | td1 lo
    g1[3] = (int)(((d1 >> 16) & 0xffffu) | (d0 << 16));             // td1 hi | tile_dim0
    g1[4] = (int)d1;                              // tile_dim1 (tile_dim2 = 0)
    g1[5] = (int)stride0;                         // tensor_dim0_stride[31:0]
    g1[6] = 0;                                    // stride0 hi | stride1 lo
    g1[7] = 0;                                    // stride1 hi
    const v4i z4 = {0, 0, 0, 0};
    const v8i z8 = {0, 0, 0, 0, 0, 0, 0, 0};
    __builtin_amdgcn_tensor_load_to_lds(g0, g1, z4, z4, z8, 0);
}
#endif

// ---------------------------------------------------------------------------
// Pre-pass 1: element-wise f32 -> bf16 (activations).
// ---------------------------------------------------------------------------
__global__ __launch_bounds__(256)
void cvt_bf16_kernel(const float* __restrict__ src, __bf16* __restrict__ dst,
                     int nelem) {
    const int i = (blockIdx.x * 256 + threadIdx.x) * 8;
    if (i < nelem) {
        v8bf o;
        #pragma unroll
        for (int j = 0; j < 8; ++j) o[j] = (__bf16)src[i + j];
        *(v8bf*)&dst[i] = o;
    }
}

// ---------------------------------------------------------------------------
// Pre-pass 2: weight transpose + convert:  Wt[c][d] = (bf16) w[d][c].
// ---------------------------------------------------------------------------
__global__ __launch_bounds__(256)
void transpose_w_kernel(const float* __restrict__ w, __bf16* __restrict__ wT) {
    __shared__ __bf16 tile[32][33];
    const int c0 = blockIdx.x * 32, d0 = blockIdx.y * 32;
    const int tx = threadIdx.x & 31, ty = threadIdx.x >> 5;  // 32 x 8
    #pragma unroll
    for (int i = ty; i < 32; i += 8)
        tile[i][tx] = (__bf16)w[(size_t)(d0 + i) * NC_ + c0 + tx];
    __syncthreads();
    #pragma unroll
    for (int i = ty; i < 32; i += 8)
        wT[(size_t)(c0 + i) * D_ + d0 + tx] = tile[tx][i];
}

// ---------------------------------------------------------------------------
// Kernel 3: projection  out[b,n,s,h] = sum_d A[b,s,d]*W[d,n,h] + bias[n,h].
// Pure vmem+wmma.  store_vt==1 stores V transposed [B][N][H][S].
// ---------------------------------------------------------------------------
__global__ __launch_bounds__(128)
void qkv_proj_kernel(const __bf16* __restrict__ A,    // [B][S][D]
                     const __bf16* __restrict__ Wt,   // [NC][D]
                     const float*  __restrict__ bias, // [NC]
                     __bf16* __restrict__ outp, int store_vt)
{
    const int tid = threadIdx.x, wave = tid >> 5, lane = tid & 31;
    const int m0 = blockIdx.x * 64, c0 = blockIdx.y * 64, b = blockIdx.z;
    const int colL = lane & 15;

    const __bf16* arow =
        A + ((size_t)b * S_ + m0 + wave * 16 + colL) * D_ + (lane >> 4) * 16;
    const __bf16* brow = Wt + (size_t)(c0 + colL) * D_ + (lane >> 4) * 16;

    v8f acc[4] = {};
    for (int d0 = 0; d0 < D_; d0 += 32) {
        ABfrag a;
        a.h[0] = *(const v8bf*)(arow + d0);
        a.h[1] = *(const v8bf*)(arow + d0 + 8);
        #pragma unroll
        for (int ct = 0; ct < 4; ++ct) {
            const v16bf bf = *(const v16bf*)(brow + (size_t)ct * 16 * D_ + d0);
            acc[ct] = __builtin_amdgcn_wmma_f32_16x16x32_bf16(
                false, a.v, false, bf, (short)0, acc[ct], false, false);
        }
    }

    const int rbase = (lane >> 4) * 8;
    #pragma unroll
    for (int ct = 0; ct < 4; ++ct) {
        const int c = c0 + ct * 16 + colL;
        const float bval = bias[c];
        const int n = c >> 6;   // c / H
        const int h = c & 63;   // c % H
        #pragma unroll
        for (int r = 0; r < 8; ++r) {
            const int s = m0 + wave * 16 + rbase + r;
            const float val = acc[ct][r] + bval;
            size_t off;
            if (store_vt) off = (((size_t)b * NH_ + n) * H_ + h) * (size_t)S_ + s;
            else          off = (((size_t)b * NH_ + n) * (size_t)S_ + s) * H_ + h;
            outp[off] = (__bf16)val;
        }
    }
}

// ---------------------------------------------------------------------------
// Kernel 4: flash-attention, double-buffered TDM pipeline.
// Block = 4 waves; wave owns 16 query rows of one (b, n).  Per 32-wide
// T-step: wait for tile i, barrier, kick DMA of tile i+1, then 4 WMMA Q*Kt,
// online softmax, P re-layout through LDS, 4 WMMA P*V.
// ---------------------------------------------------------------------------
__global__ __launch_bounds__(128)
void attn_kernel(const __bf16* __restrict__ Q,    // [B][N][F][H]
                 const __bf16* __restrict__ K,    // [B][N][T][H]
                 const __bf16* __restrict__ Vt,   // [B][N][H][T]
                 const int*    __restrict__ mask, // [B][F][T]
                 float*        __restrict__ out)  // [B][F][N][H]
{
    __shared__ __align__(64) __bf16 ldsK [2][32][64];  // [buf][t_local][h]
    __shared__ __align__(64) __bf16 ldsVt[2][64][32];  // [buf][h][t_local]
    __shared__ __align__(64) __bf16 ldsP [4][16][32];  // per-wave P re-layout

    const int tid  = threadIdx.x;
    const int wave = tid >> 5;
    const int lane = tid & 31;
    const int f0   = blockIdx.x * 64;
    const int n    = blockIdx.y;
    const int b    = blockIdx.z;

    const __bf16* Qb = Q  + ((size_t)b * NH_ + n) * (size_t)S_ * H_;
    const __bf16* Kb = K  + ((size_t)b * NH_ + n) * (size_t)S_ * H_;
    const __bf16* Vb = Vt + ((size_t)b * NH_ + n) * (size_t)H_ * S_;

    const int rbase = (lane >> 4) * 8;
    const int colL  = lane & 15;
    const int* Mrow = mask + ((size_t)b * S_ + f0 + wave * 16 + rbase) * S_ + colL;

    // stage K 32x64 and Vt 64x32 bf16 tiles for T-offset t0 into buffer pb
    auto stage_tiles = [&](int t0, int pb) {
#if defined(HAVE_TDM)
        if (tid < 32) {  // wave 0 drives the Tensor Data Mover for the block
            tdm_load_2d(Kb + (size_t)t0 * H_, &ldsK[pb][0][0],  H_, 32, H_);
            tdm_load_2d(Vb + t0,              &ldsVt[pb][0][0], 32, H_, S_);
        }
#else
        #pragma unroll
        for (int c = tid; c < 256; c += 128) {          // K: 256 x 16B chunks
            const int row = c >> 3, off = (c & 7) * 8;
            copy16_g2l(Kb + (size_t)(t0 + row) * H_ + off, &ldsK[pb][row][off]);
        }
        #pragma unroll
        for (int c = tid; c < 256; c += 128) {          // Vt: 256 x 16B chunks
            const int row = c >> 2, off = (c & 3) * 8;
            copy16_g2l(Vb + (size_t)row * S_ + t0 + off, &ldsVt[pb][row][off]);
        }
#endif
    };
    auto wait_tiles = [&]() {
#if defined(HAVE_TDM)
        if (tid < 32) __builtin_amdgcn_s_wait_tensorcnt(0);
#elif defined(HAVE_ASYNC_LDS)
#if defined(HAVE_WAIT_ASYNC)
        __builtin_amdgcn_s_wait_asynccnt(0);
#else
        asm volatile("s_wait_asynccnt 0" ::: "memory");
#endif
#endif
    };

    // --- Q fragments once: two 16x32 chunks over H ---
    ABfrag qfrag[2];
    {
        const __bf16* qrow =
            Qb + (size_t)(f0 + wave * 16 + colL) * H_ + (lane >> 4) * 16;
        #pragma unroll
        for (int hc = 0; hc < 2; ++hc) {
            qfrag[hc].h[0] = *(const v8bf*)(qrow + hc * 32);
            qfrag[hc].h[1] = *(const v8bf*)(qrow + hc * 32 + 8);
        }
    }

    float m_run[8], l_run[8];
    v8f   o_acc[4] = {};
    #pragma unroll
    for (int r = 0; r < 8; ++r) { m_run[r] = -3.0e38f; l_run[r] = 0.0f; }

    const float scale = 0.125f;  // 1/sqrt(64)

    stage_tiles(0, 0);           // prologue: tile 0 -> buffer 0

    int pb = 0;
    for (int t0 = 0; t0 < S_; t0 += 32, pb ^= 1) {
        // tile t0 is the only outstanding transfer -> exact wait
        wait_tiles();
        // one barrier: LDS visible to all waves AND all waves are done
        // reading buffer pb^1 (so it is safe to overwrite it next)
        __syncthreads();
        if (t0 + 32 < S_)
            stage_tiles(t0 + 32, pb ^ 1);   // DMA flies under the math below
        if (t0 + 32 < S_)
            __builtin_prefetch(Mrow + (size_t)(t0 + 32), 0, 1);

        // --- scores S = Q*K^T : two 16x16 col tiles, 2 H-chunks each ---
        v8f sc[2] = {};
        #pragma unroll
        for (int ct = 0; ct < 2; ++ct) {
            const int tl = ct * 16 + colL;
            #pragma unroll
            for (int hc = 0; hc < 2; ++hc) {
                const int kb = hc * 32 + (lane >> 4) * 16;
                const v16bf kfrag = *(const v16bf*)&ldsK[pb][tl][kb];
                sc[ct] = __builtin_amdgcn_wmma_f32_16x16x32_bf16(
                    false, qfrag[hc].v, false, kfrag, (short)0, sc[ct], false, false);
            }
        }

        // --- mask + scale + online softmax (rows live in 16-lane groups) ---
        #pragma unroll
        for (int r = 0; r < 8; ++r) {
            const float am0 = 1.0f - (float)Mrow[(size_t)r * S_ + t0];
            const float am1 = 1.0f - (float)Mrow[(size_t)r * S_ + t0 + 16];
            float s0 = sc[0][r] * scale + am0 * -10000.0f;
            float s1 = sc[1][r] * scale + am1 * -10000.0f;
            float mx = fmaxf(s0, s1);
            #pragma unroll
            for (int off = 1; off < 16; off <<= 1)
                mx = fmaxf(mx, __shfl_xor(mx, off, 32));
            const float mnew = fmaxf(m_run[r], mx);
            const float p0 = __expf(s0 - mnew);
            const float p1 = __expf(s1 - mnew);
            sc[0][r] = p0; sc[1][r] = p1;
            float rs = p0 + p1;
            #pragma unroll
            for (int off = 1; off < 16; off <<= 1)
                rs += __shfl_xor(rs, off, 32);
            const float corr = __expf(m_run[r] - mnew);
            l_run[r] = l_run[r] * corr + rs;
            m_run[r] = mnew;
            #pragma unroll
            for (int ct = 0; ct < 4; ++ct) o_acc[ct][r] *= corr;
        }

        // --- P: C-layout -> bf16 -> LDS -> A-layout fragment ---
        #pragma unroll
        for (int r = 0; r < 8; ++r) {
            const int fl = rbase + r;
            ldsP[wave][fl][colL]      = (__bf16)sc[0][r];
            ldsP[wave][fl][colL + 16] = (__bf16)sc[1][r];
        }
        ABfrag pfrag;
        {
            const int kb = (lane >> 4) * 16;
            pfrag.h[0] = *(const v8bf*)&ldsP[wave][colL][kb];
            pfrag.h[1] = *(const v8bf*)&ldsP[wave][colL][kb + 8];
        }
        // --- O += P*V : 4 col tiles over H ---
        #pragma unroll
        for (int ht = 0; ht < 4; ++ht) {
            const int hcol = ht * 16 + colL;
            const int kb   = (lane >> 4) * 16;
            const v16bf vfrag = *(const v16bf*)&ldsVt[pb][hcol][kb];
            o_acc[ht] = __builtin_amdgcn_wmma_f32_16x16x32_bf16(
                false, pfrag.v, false, vfrag, (short)0, o_acc[ht], false, false);
        }
    }

    // --- finalize: divide by row sum, store context [B][F][N][H] f32 ---
    #pragma unroll
    for (int ht = 0; ht < 4; ++ht) {
        const int h = ht * 16 + colL;
        #pragma unroll
        for (int r = 0; r < 8; ++r) {
            const int f = f0 + wave * 16 + rbase + r;
            out[(((size_t)b * S_ + f) * NH_ + n) * H_ + h] = o_acc[ht][r] / l_run[r];
        }
    }
}

// ---------------------------------------------------------------------------
extern "C" void kernel_launch(void* const* d_in, const int* in_sizes, int n_in,
                              void* d_out, int out_size, void* d_ws, size_t ws_size,
                              hipStream_t stream) {
    const float* from_tensor = (const float*)d_in[0];
    const float* to_tensor   = (const float*)d_in[1];
    const int*   attn_mask   = (const int*)  d_in[2];
    const float* wq = (const float*)d_in[3];
    const float* bq = (const float*)d_in[4];
    const float* wk = (const float*)d_in[5];
    const float* bk = (const float*)d_in[6];
    const float* wv = (const float*)d_in[7];
    const float* bv = (const float*)d_in[8];
    float* out = (float*)d_out;

    const size_t qkvE = (size_t)B_ * NH_ * S_ * H_;  // 4 Mi elems
    const size_t actE = (size_t)B_ * S_ * D_;        // 4 Mi elems
    const size_t wE   = (size_t)D_ * NC_;            // 1 Mi elems

    __bf16* qws  = (__bf16*)d_ws;          // workspace: 46 MB total (bf16)
    __bf16* kws  = qws  + qkvE;
    __bf16* vws  = kws  + qkvE;
    __bf16* abf  = vws  + qkvE;            // from_tensor bf16
    __bf16* tbf  = abf  + actE;            // to_tensor bf16
    __bf16* wtq  = tbf  + actE;            // Wq^T bf16 [NC][D]
    __bf16* wtk  = wtq  + wE;
    __bf16* wtv  = wtk  + wE;

    // pre-passes
    cvt_bf16_kernel<<<dim3((unsigned)(actE / 8 / 256)), dim3(256), 0, stream>>>(
        from_tensor, abf, (int)actE);
    cvt_bf16_kernel<<<dim3((unsigned)(actE / 8 / 256)), dim3(256), 0, stream>>>(
        to_tensor, tbf, (int)actE);
    dim3 gt(NC_ / 32, D_ / 32);
    transpose_w_kernel<<<gt, dim3(256), 0, stream>>>(wq, wtq);
    transpose_w_kernel<<<gt, dim3(256), 0, stream>>>(wk, wtk);
    transpose_w_kernel<<<gt, dim3(256), 0, stream>>>(wv, wtv);

    // projections
    dim3 blk(128);
    dim3 g1(S_ / 64, NC_ / 64, B_);
    qkv_proj_kernel<<<g1, blk, 0, stream>>>(abf, wtq, bq, qws, 0);
    qkv_proj_kernel<<<g1, blk, 0, stream>>>(tbf, wtk, bk, kws, 0);
    qkv_proj_kernel<<<g1, blk, 0, stream>>>(tbf, wtv, bv, vws, 1);

    // attention
    dim3 g2(S_ / 64, NH_, B_);
    attn_kernel<<<g2, blk, 0, stream>>>(qws, kws, vws, attn_mask, out);
}
